// ProtoConv2d_67877663146264
// MI455X (gfx1250) — matrix operand order repaired and checked by reading, not compiled, not run on hardware
//
#include <hip/hip_runtime.h>

#define HH   56
#define LL   3136          // 56*56
#define PP   576           // 64*3*3
#define MM   256           // num centers
#define CIN  64
#define COUT 128
#define NB   32
#define NN   (NB * LL)     // 100352 patch rows
#define ROWS 32            // patch rows per block (3136 % 32 == 0 -> no batch straddle)

#define A_STRIDE_U32 292   // bf16 A-tile row stride in dwords (584 bf16), 16B aligned, low-conflict
#define S_STRIDE_U32 132   // bf16 softmax tile row stride in dwords (264 bf16)

typedef __attribute__((ext_vector_type(16))) __bf16 v16bf;
typedef __attribute__((ext_vector_type(8)))  float  v8f;
typedef __attribute__((ext_vector_type(4)))  unsigned int u32x4;

union BfVec {
    v16bf v;
    u32x4 q[2];
    unsigned u32[8];
    unsigned short us[16];
};

__device__ __forceinline__ unsigned short f2bf(float f) {
    unsigned u = __float_as_uint(f);
    u += 0x7FFFu + ((u >> 16) & 1u);   // round-to-nearest-even
    return (unsigned short)(u >> 16);
}
__device__ __forceinline__ float bf2f(unsigned short s) {
    return __uint_as_float(((unsigned)s) << 16);
}

// branch-free im2col element: clamp address, select zero (no exec divergence)
__device__ __forceinline__ float patch_val(const float* __restrict__ xb,
                                           int oy, int ox, int k) {
    int c = k / 9;
    int rem = k - c * 9;
    int ki = rem / 3;
    int kj = rem - ki * 3;
    int y = oy + ki - 1;
    int z = ox + kj - 1;
    bool ok = (y >= 0) && (y < HH) && (z >= 0) && (z < HH);
    int yc = y < 0 ? 0 : (y > HH - 1 ? HH - 1 : y);
    int zc = z < 0 ? 0 : (z > HH - 1 ? HH - 1 : z);
    float v = xb[c * LL + yc * HH + zc];
    return ok ? v : 0.0f;
}

// ---------------- prep: bf16 conversions / transposes ----------------
__global__ void prep_kernel(const float* __restrict__ centers,
                            const float* __restrict__ weight,
                            unsigned short* __restrict__ cbf,   // (m,p) bf16
                            unsigned short* __restrict__ cbT,   // (p,m) bf16
                            unsigned short* __restrict__ wB) {  // (n,k) bf16
    int gid = blockIdx.x * blockDim.x + threadIdx.x;
    if (gid < MM * PP) {
        cbf[gid] = f2bf(centers[gid]);
        int p = gid >> 8, m = gid & 255;           // PP*MM indexed as p*256+m
        cbT[gid] = f2bf(centers[m * PP + p]);
    }
    if (gid < COUT * PP) wB[gid] = f2bf(weight[gid]);
}

// ---------------- center squared norms ----------------
__global__ void c2_kernel(const float* __restrict__ centers, float* __restrict__ c2) {
    __shared__ float red[PP];
    int m = blockIdx.x, t = threadIdx.x;
    float v = centers[m * PP + t];
    red[t] = v * v;
    __syncthreads();
    for (int s = 288; s >= 9; s >>= 1) {
        if (t < s) red[t] += red[t + s];
        __syncthreads();
    }
    if (t == 0) {
        float sum = 0.f;
        for (int i = 0; i < 9; ++i) sum += red[i];
        c2[m] = sum;
    }
}

// ---------------- fused: cdist GEMM + softmax + transform GEMM + blend ----------------
// 256 threads = 8 waves; 32 patch rows x 256 centers per block.
// Wave w: row-half rh = w&1, col-group cg = w>>1 (64 centers / 144 dims).
__global__ void __launch_bounds__(256)
fused_assign_kernel(const float* __restrict__ x,
                    const unsigned* __restrict__ cbf_u,   // centers bf16 (m,p) as dwords
                    const unsigned* __restrict__ cbT_u,   // centers bf16 (p,m) as dwords
                    const float* __restrict__ c2,
                    const float* __restrict__ temp,
                    float* __restrict__ finalw) {
    __shared__ u32x4 AbfQ[ROWS * (A_STRIDE_U32 / 4)];   // 37.4 KB: A tile bf16
    __shared__ u32x4 SbfQ[ROWS * (S_STRIDE_U32 / 4)];   // 16.9 KB: logits/softmax bf16
    __shared__ float red2[256];
    __shared__ float rowstat[ROWS];
    __shared__ float a2row[ROWS];

    unsigned* Abf_u = (unsigned*)AbfQ;
    unsigned short* Abf_s = (unsigned short*)AbfQ;
    unsigned* Sbf_u = (unsigned*)SbfQ;
    unsigned short* Sbf_s = (unsigned short*)SbfQ;

    const int tid  = threadIdx.x;
    const int lane = tid & 31;
    const int wv   = tid >> 5;
    const int row0 = blockIdx.x * ROWS;
    const int bblk = row0 / LL;
    const int l0   = row0 - bblk * LL;
    const float tempv = temp[0];
    const float* xb = x + (size_t)bblk * CIN * LL;

    // ---- phase 0: cooperative im2col -> bf16 LDS tile, fused row norms ----
    // m = tid&31 (row-fastest): consecutive lanes load consecutive l -> coalesced.
    {
        int m = tid & 31, kg = tid >> 5;       // 8 channel-groups
        int lm = l0 + m;
        int oy = lm / HH, ox = lm - oy * HH;
        float ssq = 0.f;
        for (int c = kg; c < CIN; c += 8) {
            const float* xc = xb + c * LL;
            #pragma unroll
            for (int r = 0; r < 9; ++r) {      // ki,kj compile-time
                int ki = r / 3, kj = r % 3;
                int y = oy + ki - 1, z = ox + kj - 1;
                bool ok = (y >= 0) && (y < HH) && (z >= 0) && (z < HH);
                int yc = y < 0 ? 0 : (y > HH - 1 ? HH - 1 : y);
                int zc = z < 0 ? 0 : (z > HH - 1 ? HH - 1 : z);
                float v = xc[yc * HH + zc];
                v = ok ? v : 0.f;
                Abf_s[m * (2 * A_STRIDE_U32) + c * 9 + r] = f2bf(v);
                ssq += v * v;
            }
        }
        red2[tid] = ssq;                       // tid == kg*32 + m
    }
    __syncthreads();
    if (tid < ROWS) {
        float s = 0.f;
        #pragma unroll
        for (int kg = 0; kg < 8; ++kg) s += red2[kg * 32 + tid];
        a2row[tid] = s;
    }
    __syncthreads();

    const int rh    = wv & 1;                  // row half
    const int cg    = wv >> 1;                 // column group
    const int mrow  = lane & 15;
    const int arow  = rh * 16 + mrow;
    const int halfA = (lane >= 16) ? 8 : 0;    // dword offset (A layout halves)
    const int halfB = (lane >= 16) ? 8 : 0;    // dword offset (B layout halves)
    const int half8 = (lane >= 16) ? 8 : 0;

    // ---- GEMM1: dot(patch, center), K = 576 ----
    v8f acc[4] = {};
    #pragma unroll 1
    for (int k0 = 0; k0 < PP; k0 += 32) {
        BfVec a;
        const unsigned* pa = Abf_u + arow * A_STRIDE_U32 + (k0 >> 1) + halfA;
        a.q[0] = *(const u32x4*)(pa);
        a.q[1] = *(const u32x4*)(pa + 4);
        #pragma unroll
        for (int t = 0; t < 4; ++t) {
            const unsigned* pb = cbf_u + (size_t)(cg * 64 + t * 16 + mrow) * 288 + (k0 >> 1) + halfB;
            BfVec bm;
            bm.q[0] = *(const u32x4*)(pb);
            bm.q[1] = *(const u32x4*)(pb + 4);
            acc[t] = __builtin_amdgcn_wmma_f32_16x16x32_bf16(
                false, a.v, false, bm.v, (short)0, acc[t], false, false);
        }
    }

    // ---- logits = -temp * sqrt(a2 + c2 - 2ab) -> bf16 LDS ----
    #pragma unroll
    for (int t = 0; t < 4; ++t) {
        int ncl = cg * 64 + t * 16 + mrow;
        float cc = c2[ncl];
        #pragma unroll
        for (int j = 0; j < 8; ++j) {
            int m = rh * 16 + j + half8;
            float d2 = a2row[m] + cc - 2.0f * acc[t][j];
            float d  = sqrtf(fmaxf(d2, 1e-12f));
            Sbf_s[m * (2 * S_STRIDE_U32) + ncl] = f2bf(-d * tempv);
        }
    }
    __syncthreads();

    // ---- softmax over the 256 columns of each of the 32 rows (bf16 in LDS) ----
    {
        int r = tid >> 3, sub = tid & 7;       // 8 threads per row
        unsigned short* Srow = Sbf_s + r * (2 * S_STRIDE_U32);
        float mx = -3.0e38f;
        for (int c = sub; c < MM; c += 8) mx = fmaxf(mx, bf2f(Srow[c]));
        red2[r * 8 + sub] = mx;
        __syncthreads();
        if (sub == 0) {
            float m2 = red2[r * 8];
            for (int i = 1; i < 8; ++i) m2 = fmaxf(m2, red2[r * 8 + i]);
            rowstat[r] = m2;
        }
        __syncthreads();
        float rm = rowstat[r];
        float sm = 0.f;
        for (int c = sub; c < MM; c += 8) {
            float e = __expf(bf2f(Srow[c]) - rm);
            Srow[c] = f2bf(e);
            sm += e;
        }
        red2[r * 8 + sub] = sm;
        __syncthreads();
        if (sub == 0) {
            float s2 = 0.f;
            for (int i = 0; i < 8; ++i) s2 += red2[r * 8 + i];
            rowstat[r] = 1.0f / s2;
        }
        __syncthreads();
        float inv = rowstat[r];
        for (int c = sub; c < MM; c += 8) Srow[c] = f2bf(bf2f(Srow[c]) * inv);
    }
    __syncthreads();

    // ---- GEMM2: transformed = S(32x256) @ centers(256x576); blend; store ----
    const float inv_tp1 = 1.0f / (tempv + 1.0f);
    #pragma unroll 1
    for (int tt = 0; tt < 9; ++tt) {             // 4 col-groups * 9 tiles * 16 = 576 dims
        const int p0 = cg * 144 + tt * 16;
        v8f acc2 = {};
        #pragma unroll 1
        for (int k0 = 0; k0 < MM; k0 += 32) {
            BfVec a;
            const unsigned* pa = Sbf_u + arow * S_STRIDE_U32 + (k0 >> 1) + halfA;
            a.q[0] = *(const u32x4*)(pa);
            a.q[1] = *(const u32x4*)(pa + 4);
            BfVec bm;
            const unsigned* pb = cbT_u + (size_t)(p0 + mrow) * 128 + (k0 >> 1) + halfB;
            bm.q[0] = *(const u32x4*)(pb);
            bm.q[1] = *(const u32x4*)(pb + 4);
            acc2 = __builtin_amdgcn_wmma_f32_16x16x32_bf16(
                false, a.v, false, bm.v, (short)0, acc2, false, false);
        }
        const int p = p0 + mrow;
        #pragma unroll
        for (int j = 0; j < 8; ++j) {
            int m  = rh * 16 + j + half8;
            int lm = l0 + m;
            int oy2 = lm / HH, ox2 = lm - oy2 * HH;
            float flat = patch_val(xb, oy2, ox2, p);        // f32 fidelity for blend
            float fin  = (tempv * acc2[j] + flat) * inv_tp1;
            finalw[(size_t)(row0 + m) * PP + p] = fin;
        }
    }
}

// ---------------- conv GEMM: fold+conv collapsed to an indexed gather ----------------
// 256 threads = 8 waves; 32 rows x 128 Cout per block.
__global__ void __launch_bounds__(256)
conv_kernel(const float* __restrict__ finalw,
            const unsigned* __restrict__ wB_u,   // weight bf16 (n,k) as dwords
            const float* __restrict__ bias,
            float* __restrict__ out) {
    __shared__ u32x4 AbfQ[ROWS * (A_STRIDE_U32 / 4)];
    unsigned* Abf_u = (unsigned*)AbfQ;
    unsigned short* Abf_s = (unsigned short*)AbfQ;

    const int tid  = threadIdx.x;
    const int lane = tid & 31;
    const int wv   = tid >> 5;
    const int row0 = blockIdx.x * ROWS;
    const int b    = row0 / LL;
    const int l0   = row0 - b * LL;

    // ---- phase 0: gather A tile: A[m][k] = final[b,k,l] with border-zero mask ----
    {
        int m = tid & 31, kg = tid >> 5;
        int lm = l0 + m;
        int oy = lm / HH, ox = lm - oy * HH;
        const float* base = finalw + (size_t)b * PP * LL + lm;
        for (int c = kg; c < CIN; c += 8) {
            #pragma unroll
            for (int r = 0; r < 9; ++r) {       // ry,rx compile-time
                int ry = r / 3, rx = r % 3;
                bool zm = ((oy == 0) && (ry == 0)) || ((ox == 0) && (rx == 0));
                int k = c * 9 + r;
                float v = base[(size_t)k * LL]; // always a valid address
                v = zm ? 0.f : v;
                Abf_s[m * (2 * A_STRIDE_U32) + k] = f2bf(v);
            }
        }
    }
    __syncthreads();

    const int rh    = wv & 1;
    const int cg    = wv >> 1;                  // 4 groups * 32 Cout
    const int mrow  = lane & 15;
    const int arow  = rh * 16 + mrow;
    const int halfA = (lane >= 16) ? 8 : 0;
    const int halfB = (lane >= 16) ? 8 : 0;
    const int half8 = (lane >= 16) ? 8 : 0;
    const int col0  = cg * 32 + mrow;

    v8f acc[2] = {};
    #pragma unroll 1
    for (int k0 = 0; k0 < PP; k0 += 32) {
        BfVec a;
        const unsigned* pa = Abf_u + arow * A_STRIDE_U32 + (k0 >> 1) + halfA;
        a.q[0] = *(const u32x4*)(pa);
        a.q[1] = *(const u32x4*)(pa + 4);
        #pragma unroll
        for (int t = 0; t < 2; ++t) {
            const unsigned* pb = wB_u + (size_t)(col0 + t * 16) * 288 + (k0 >> 1) + halfB;
            BfVec bm;
            bm.q[0] = *(const u32x4*)(pb);
            bm.q[1] = *(const u32x4*)(pb + 4);
            acc[t] = __builtin_amdgcn_wmma_f32_16x16x32_bf16(
                false, a.v, false, bm.v, (short)0, acc[t], false, false);
        }
    }
    #pragma unroll
    for (int t = 0; t < 2; ++t) {
        int col = col0 + t * 16;
        float bz = bias[col];
        size_t obase = (size_t)b * (COUT * LL) + (size_t)col * LL + l0 + rh * 16 + half8;
        #pragma unroll
        for (int j = 0; j < 8; ++j) out[obase + j] = acc[t][j] + bz;
    }
}

extern "C" void kernel_launch(void* const* d_in, const int* in_sizes, int n_in,
                              void* d_out, int out_size, void* d_ws, size_t ws_size,
                              hipStream_t stream) {
    const float* x       = (const float*)d_in[0];
    const float* weight  = (const float*)d_in[1];
    const float* bias    = (const float*)d_in[2];
    const float* centers = (const float*)d_in[3];
    const float* temp    = (const float*)d_in[4];
    float* out = (float*)d_out;

    char* ws = (char*)d_ws;
    size_t off = 0;
    float* finalw = (float*)(ws + off);                 off += (size_t)NN * PP * sizeof(float);
    unsigned short* cbf = (unsigned short*)(ws + off);  off += (size_t)MM * PP * sizeof(unsigned short);
    unsigned short* cbT = (unsigned short*)(ws + off);  off += (size_t)MM * PP * sizeof(unsigned short);
    unsigned short* wB  = (unsigned short*)(ws + off);  off += (size_t)PP * COUT * sizeof(unsigned short);
    float* c2 = (float*)(ws + off);                     off += (size_t)MM * sizeof(float);
    (void)ws_size; (void)in_sizes; (void)n_in; (void)out_size;

    prep_kernel<<<(MM * PP + 255) / 256, 256, 0, stream>>>(centers, weight, cbf, cbT, wB);
    c2_kernel<<<MM, PP, 0, stream>>>(centers, c2);
    fused_assign_kernel<<<NN / ROWS, 256, 0, stream>>>(
        x, (const unsigned*)cbf, (const unsigned*)cbT, c2, temp, finalw);
    conv_kernel<<<NN / ROWS, 256, 0, stream>>>(finalw, (const unsigned*)wB, bias, out);
}
